// GATv2Layer_40862318854262
// MI455X (gfx1250) — compile-verified, matrix operand order checked by reading, not run on hardware
//
#include <hip/hip_runtime.h>
#include <stdint.h>

typedef __attribute__((ext_vector_type(16))) __bf16 v16bf;
typedef __attribute__((ext_vector_type(8)))  float  v8f;
typedef __attribute__((ext_vector_type(4)))  float  vf4;
typedef __attribute__((ext_vector_type(2)))  float  vf2;

#define LH_STRIDE 132   // 128 pairs/row + 4 pad (bank spread)
#define G_STRIDE  260   // 256 floats/row + 4 pad (bank spread)

__device__ __forceinline__ uint32_t f2bf_pack(float lo, float hi) {
    uint32_t ul = __float_as_uint(lo);
    uint32_t uh = __float_as_uint(hi);
    ul = (ul + 0x7FFFu + ((ul >> 16) & 1u)) >> 16;   // RNE f32 -> bf16
    uh = (uh + 0x7FFFu + ((uh >> 16) & 1u)) >> 16;
    return (ul & 0xFFFFu) | (uh << 16);
}

union FragA { uint32_t u[8]; v16bf v; };
union FragB { uint4 q[2];    v16bf v; };

// Repack W_l / W_r (fp32, DxO row-major) into WMMA B-fragment-native bf16 pairs.
// idx = ((((p*16+ct)*8)+kt)*32 + lane)*8 + v  -> one uint32 (two bf16, K=k,k+1)
// where col c = ct*16 + (lane&15), k = kt*32 + (lane>>4)*8 + (v<4 ? 2v : 16+2(v-4)).
__global__ __launch_bounds__(256) void gat_pack_w(const float* __restrict__ Wl,
                                                  const float* __restrict__ Wr,
                                                  uint32_t* __restrict__ wp) {
    int tid  = blockIdx.x * 256 + threadIdx.x;        // 65536 total
    int v    = tid & 7;
    int lane = (tid >> 3) & 31;
    int kt   = (tid >> 8) & 7;
    int ct   = (tid >> 11) & 15;
    int p    = (tid >> 15) & 1;
    int c    = ct * 16 + (lane & 15);
    int koff = ((lane >> 4) * 8) + (v < 4 ? 2 * v : 16 + 2 * (v - 4));
    int k    = kt * 32 + koff;
    const float* W = p ? Wr : Wl;
    wp[tid] = f2bf_pack(W[k * 256 + c], W[(k + 1) * 256 + c]);
}

__global__ __launch_bounds__(256) void gatv2_fused(const float* __restrict__ h,
                                                   const float* __restrict__ eb,
                                                   const float* __restrict__ a,
                                                   const float* __restrict__ gamma,
                                                   const float* __restrict__ beta,
                                                   const uint32_t* __restrict__ wp,
                                                   float* __restrict__ out) {
    __shared__ uint32_t lh32[16 * LH_STRIDE];   // h tile, bf16 pairs
    __shared__ float    gbuf[2][16 * G_STRIDE]; // g_l, g_r (fp32)
    __shared__ float    a_s[256];
    __shared__ float    eb_s[16];
    __shared__ float    e_s[256];
    __shared__ float    alpha_s[256];

    const int t    = threadIdx.x;
    const int lane = t & 31;
    const int wave = t >> 5;
    const int r0   = blockIdx.x * 16;           // 16 rows = 4 batch elems

    // ---- stage h (fp32 -> packed bf16 in LDS), plus a / edge_bias ----
    #pragma unroll
    for (int q = 0; q < 8; ++q) {
        int p   = t + q * 256;                  // pair index 0..2047
        int row = p >> 7, kp = p & 127;
        const vf2 hv = __builtin_nontemporal_load(
            (const vf2*)(h + (size_t)(r0 + row) * 256 + kp * 2));
        lh32[row * LH_STRIDE + kp] = f2bf_pack(hv.x, hv.y);
    }
    a_s[t] = a[t];                               // (H=4, d=64) flat
    if (t < 16) eb_s[t] = eb[t];                 // (N=4, N=4)
    __syncthreads();

    // ---- load full A fragment (16x256, all 8 K-chunks) once: 64 VGPRs ----
    const int arow = lane & 15;
    const int klo2 = (lane >> 4) * 4;            // half-wave K split (pair units)
    FragA Af[8];
    #pragma unroll
    for (int kt = 0; kt < 8; ++kt) {
        const int kb2 = kt * 16 + klo2;
        #pragma unroll
        for (int v = 0; v < 8; ++v)
            Af[kt].u[v] = lh32[arow * LH_STRIDE + kb2 + (v < 4 ? v : 4 + v)];
    }

    // ---- projections: 32 tiles (p=0: g_l, p=1: g_r), 4 tiles per wave ----
    // Literal trip count => no divergent-loop exec machinery around WMMA.
    const int bcol = lane & 15;
    const int rb   = (lane >> 4) * 8;            // C-frag row base
    #pragma unroll
    for (int tt = 0; tt < 4; ++tt) {
        const int T = wave * 4 + tt;
        const int p = T >> 4, ct = T & 15;
        v8f acc = {0.f, 0.f, 0.f, 0.f, 0.f, 0.f, 0.f, 0.f};
        const uint32_t* wb = wp + (size_t)((p * 16 + ct) * 8) * 256 + lane * 8;
        #pragma unroll
        for (int kt = 0; kt < 8; ++kt) {
            FragB B;                             // 2 x global_load_b128 (L2-resident)
            const uint32_t* wk = wb + kt * 256;
            B.q[0] = *(const uint4*)(wk);
            B.q[1] = *(const uint4*)(wk + 4);
            acc = __builtin_amdgcn_wmma_f32_16x16x32_bf16(
                false, Af[kt].v, false, B.v, (short)0, acc, false, false);
        }
        float* g = gbuf[p];
        #pragma unroll
        for (int r = 0; r < 8; ++r)
            g[(rb + r) * G_STRIDE + ct * 16 + bcol] = acc[r];
    }
    __syncthreads();

    // ---- e[b,i,j,h]: one (bb,i,j,head) per thread ----
    {
        const int bb = (t >> 6) & 3, i = (t >> 4) & 3, j = (t >> 2) & 3, hh = t & 3;
        const float* glr = &gbuf[0][(bb * 4 + i) * G_STRIDE];
        const float* grr = &gbuf[1][(bb * 4 + j) * G_STRIDE];
        float s = 0.f;
        #pragma unroll 4
        for (int ttc = 0; ttc < 64; ++ttc) {
            const int c = hh * 64 + ttc;
            float x = glr[c] + grr[c];
            x = (x > 0.f) ? x : 0.2f * x;        // LeakyReLU(0.2) before scoring
            s = fmaf(x, a_s[c], s);
        }
        e_s[t] = s + eb_s[i * 4 + j];
    }
    __syncthreads();

    // ---- softmax over j (4 neighbors) ----
    {
        const int base = t & ~12;
        float e0 = e_s[base], e1 = e_s[base | 4], e2 = e_s[base | 8], e3 = e_s[base | 12];
        float m  = fmaxf(fmaxf(e0, e1), fmaxf(e2, e3));
        float d  = __expf(e0 - m) + __expf(e1 - m) + __expf(e2 - m) + __expf(e3 - m);
        alpha_s[t] = __expf(e_s[t] - m) / d;
    }
    __syncthreads();

    // ---- h' = sum_j alpha * g_r, then LayerNorm over 256 channels ----
    {
        const int row  = t >> 4;                 // 0..15 -> (bb, i)
        const int q    = t & 15;
        const int c0   = q * 16;                 // 16-channel chunk (head-aligned)
        const int head = q >> 2;
        const int bb   = row >> 2;
        float al[4];
        #pragma unroll
        for (int j = 0; j < 4; ++j) al[j] = alpha_s[(row * 4 + j) * 4 + head];
        float hp[16]; float s1 = 0.f, s2 = 0.f;
        #pragma unroll
        for (int cc = 0; cc < 16; ++cc) {
            const int c = c0 + cc;
            float v = al[0] * gbuf[1][(bb * 4 + 0) * G_STRIDE + c]
                    + al[1] * gbuf[1][(bb * 4 + 1) * G_STRIDE + c]
                    + al[2] * gbuf[1][(bb * 4 + 2) * G_STRIDE + c]
                    + al[3] * gbuf[1][(bb * 4 + 3) * G_STRIDE + c];
            hp[cc] = v; s1 += v; s2 += v * v;
        }
        #pragma unroll
        for (int msk = 1; msk < 16; msk <<= 1) { // row reduce: 16 lanes share a row
            s1 += __shfl_xor(s1, msk, 32);
            s2 += __shfl_xor(s2, msk, 32);
        }
        const float mean = s1 * (1.f / 256.f);
        const float var  = s2 * (1.f / 256.f) - mean * mean;
        const float inv  = rsqrtf(var + 1e-5f);
        float* op = out + (size_t)(r0 + row) * 256 + c0;
        #pragma unroll
        for (int v4 = 0; v4 < 4; ++v4) {         // 4 x NT b128 stores, coalesced
            const vf4 gm = *(const vf4*)(gamma + c0 + v4 * 4);
            const vf4 bt = *(const vf4*)(beta  + c0 + v4 * 4);
            vf4 o;
            o.x = (hp[v4 * 4 + 0] - mean) * inv * gm.x + bt.x;
            o.y = (hp[v4 * 4 + 1] - mean) * inv * gm.y + bt.y;
            o.z = (hp[v4 * 4 + 2] - mean) * inv * gm.z + bt.z;
            o.w = (hp[v4 * 4 + 3] - mean) * inv * gm.w + bt.w;
            __builtin_nontemporal_store(o, (vf4*)(op + v4 * 4));
        }
    }
}

extern "C" void kernel_launch(void* const* d_in, const int* in_sizes, int n_in,
                              void* d_out, int out_size, void* d_ws, size_t ws_size,
                              hipStream_t stream) {
    const float* h     = (const float*)d_in[0];
    const float* eb    = (const float*)d_in[1];
    const float* Wl    = (const float*)d_in[2];
    const float* Wr    = (const float*)d_in[3];
    const float* a     = (const float*)d_in[4];
    const float* gamma = (const float*)d_in[5];
    const float* beta  = (const float*)d_in[6];
    float*    out = (float*)d_out;
    uint32_t* wp  = (uint32_t*)d_ws;             // 256 KB packed bf16 weights

    const int B = in_sizes[0] / (4 * 256);       // 32768

    gat_pack_w<<<256, 256, 0, stream>>>(Wl, Wr, wp);
    gatv2_fused<<<B / 4, 256, 0, stream>>>(h, eb, a, gamma, beta, wp, out);
}
